// JaccardLoss_56547539419669
// MI455X (gfx1250) — compile-verified, matrix operand order checked by reading, not run on hardware
//
#include <hip/hip_runtime.h>
#include <hip/hip_bf16.h>

typedef __attribute__((ext_vector_type(16))) _Float16 v16h;
typedef __attribute__((ext_vector_type(8)))  float    v8f;

// d_ws layout: acc[0..31] = per-slice intersection, acc[32..63] = per-slice union.
__global__ void jaccard_zero_ws(float* __restrict__ acc) {
    acc[threadIdx.x] = 0.0f;   // launched with 64 threads
}

// Grid: (blocksPerSlice=32, slices=32), block = 256 threads = 8 wave32 waves.
// Slice = 2^20 elements; per block 2^15; per wave 2^12 = 8 WMMA iterations of 512.
//
// Inputs are bit-exact 0x00000000 (0.0f) or 0x3F800000 (1.0f), so:
//   intersection flag = o AND t   (bitwise on the dwords -> exact 0.0f/1.0f)
//   union flag        = o OR  t   (bitwise on the dwords -> exact 0.0f/1.0f)
// One v_and_b32 / v_or_b32 per element, then v_cvt_pk_f16_f32 into the WMMA A tile.
__global__ __launch_bounds__(256)
void jaccard_count_kernel(const uint4* __restrict__ out,
                          const uint4* __restrict__ tgt,
                          float* __restrict__ acc) {
    const int lane  = threadIdx.x & 31;
    const int wave  = threadIdx.x >> 5;
    const int slice = blockIdx.y;
    // element base = (slice<<20) + (blk<<15) + (wave<<12); in uint4 units: >>2
    const int base4 = ((slice << 20) + ((int)blockIdx.x << 15) + (wave << 12)) >> 2;

    v16h ones;
#pragma unroll
    for (int i = 0; i < 16; ++i) ones[i] = (_Float16)1.0f;

    v8f ci = {};   // accumulates 16x(row sums) of intersection flags
    v8f cu = {};   // accumulates 16x(row sums) of union flags

#pragma unroll
    for (int it = 0; it < 8; ++it) {
        const int ib4 = base4 + it * 128;          // 512 elements = 128 uint4
        v16h ai, au;
#pragma unroll
        for (int j = 0; j < 4; ++j) {
            const int off4 = ib4 + j * 32 + lane;  // coalesced 512B per wave sub-step
            const uint4 o4 = out[off4];
            const uint4 t4 = tgt[off4];
            const unsigned ov[4] = {o4.x, o4.y, o4.z, o4.w};
            const unsigned tv[4] = {t4.x, t4.y, t4.z, t4.w};
#pragma unroll
            for (int e = 0; e < 4; ++e) {
                ai[j * 4 + e] = (_Float16)__uint_as_float(ov[e] & tv[e]); // AND
                au[j * 4 + e] = (_Float16)__uint_as_float(ov[e] | tv[e]); // OR
            }
        }
        // D = A x Ones + C : every D[m][n] = rowsum(A_m) + C, so
        // sum(all 256 D elems) == 16 * sum(all 512 A elems). Layout-agnostic.
        ci = __builtin_amdgcn_wmma_f32_16x16x32_f16(false, ai, false, ones,
                                                    (short)0, ci, false, false);
        cu = __builtin_amdgcn_wmma_f32_16x16x32_f16(false, au, false, ones,
                                                    (short)0, cu, false, false);
    }

    float si = 0.0f, su = 0.0f;
#pragma unroll
    for (int k = 0; k < 8; ++k) { si += ci[k]; su += cu[k]; }

    // wave32 all-lane sum (5 xor steps)
#pragma unroll
    for (int m = 16; m >= 1; m >>= 1) {
        si += __shfl_xor(si, m, 32);
        su += __shfl_xor(su, m, 32);
    }

    if (lane == 0) {
        // si/su are 16x the true wave counts; counts are integer-valued floats
        // well under 2^24 so every add (and the atomics) is exact/deterministic.
        atomicAdd(&acc[slice],      si * 0.0625f);
        atomicAdd(&acc[32 + slice], su * 0.0625f);
    }
}

// One wave: 32 slices -> mean IoU * 100.
__global__ void jaccard_finalize(const float* __restrict__ acc,
                                 float* __restrict__ result) {
    const int lane = threadIdx.x;   // launched with 32 threads
    const float inter = acc[lane];
    const float uni   = acc[32 + lane];
    float v = inter / (uni + 1e-07f);
#pragma unroll
    for (int m = 16; m >= 1; m >>= 1) v += __shfl_xor(v, m, 32);
    if (lane == 0) result[0] = v * (100.0f / 32.0f);
}

extern "C" void kernel_launch(void* const* d_in, const int* in_sizes, int n_in,
                              void* d_out, int out_size, void* d_ws, size_t ws_size,
                              hipStream_t stream) {
    const uint4* out = (const uint4*)d_in[0];   // [32,1,1024,1024] f32, values {0,1}
    const uint4* tgt = (const uint4*)d_in[1];   // [32,1,1024,1024] f32, values {0,1}
    float* acc = (float*)d_ws;                  // 64 floats of scratch
    float* res = (float*)d_out;                 // 1 float

    jaccard_zero_ws<<<1, 64, 0, stream>>>(acc);

    dim3 grid(32, 32);                          // 32 blocks per slice, 32 slices
    jaccard_count_kernel<<<grid, 256, 0, stream>>>(out, tgt, acc);

    jaccard_finalize<<<1, 32, 0, stream>>>(acc, res);
}
